// TransformerMusicModel_13752485282215
// MI455X (gfx1250) — matrix operand order repaired and checked by reading, not compiled, not run on hardware
//
#include <hip/hip_runtime.h>
#include <hip/hip_bf16.h>

// Problem constants (match reference)
#define BB   4
#define SS   2048
#define DD   768
#define HH   12
#define LLN  6
#define FFD  3072
#define VV   512
#define HDIM 64
#define MTOT (BB*SS)   // 8192 token rows

typedef __bf16 bf16_t;
typedef __attribute__((ext_vector_type(16))) __bf16 v16bf;
typedef __attribute__((ext_vector_type(8)))  float  v8f;

union FragU  { v16bf v; uint4 u[2]; };
union Pack8  { uint4 u; unsigned short us[8]; };

// Detect gfx1250 async global->LDS builtins (fallback: staged VGPR copy)
#if defined(__has_builtin)
#  if __has_builtin(__builtin_amdgcn_global_load_async_to_lds_b128) && \
      __has_builtin(__builtin_amdgcn_s_wait_asynccnt)
#    define USE_ASYNC_LDS 1
#  endif
#endif
#ifndef USE_ASYNC_LDS
#  define USE_ASYNC_LDS 0
#endif

#if USE_ASYNC_LDS
// Builtin signature (from clang diagnostics):
//   (int __vector(4) addrspace(1)* gsrc, int __vector(4) addrspace(3)* ldst,
//    imm int offset, imm int cpol)
typedef int v4i_as __attribute__((vector_size(16)));
typedef __attribute__((address_space(1))) v4i_as* gptr_b128_t;
typedef __attribute__((address_space(3))) v4i_as* lptr_b128_t;
#define ASYNC_COPY_B128(gsrc, ldst)                                          \
  __builtin_amdgcn_global_load_async_to_lds_b128(                            \
      (gptr_b128_t)(gsrc), (lptr_b128_t)(ldst), 0, 0)
#define ASYNC_WAIT() __builtin_amdgcn_s_wait_asynccnt(0)
#endif

// ---------------------------------------------------------------------------
// WMMA fragment loaders (bf16 16x16x32). Source region must hold 32
// contiguous K-values (K-major LDS/global layout).
// A-matrix (16x32): lane holds row = lane%16, K = {lhi*8..+7, 16+lhi*8..+7}
// B-matrix (32x16): lane holds col = lane%16, K = {lhi*8..+7, 16+lhi*8..+7}
// ---------------------------------------------------------------------------
__device__ __forceinline__ v16bf load_frag_a(const bf16_t* base, int lane) {
  FragU f;
  const int h = (lane >> 4) * 8;
  f.u[0] = *(const uint4*)(base + h);
  f.u[1] = *(const uint4*)(base + 16 + h);
  return f.v;
}
__device__ __forceinline__ v16bf load_frag_b(const bf16_t* base, int lane) {
  FragU f;
  const int h = (lane >> 4) * 8;
  f.u[0] = *(const uint4*)(base + h);
  f.u[1] = *(const uint4*)(base + 16 + h);
  return f.v;
}
__device__ __forceinline__ v8f wmma_bf16(v16bf a, v16bf b, v8f c) {
  return __builtin_amdgcn_wmma_f32_16x16x32_bf16(false, a, false, b,
                                                 (short)0, c, false, false);
}

// ---------------------------------------------------------------------------
// fp32 -> bf16 bulk convert
// ---------------------------------------------------------------------------
__global__ __launch_bounds__(256) void cvt_bf16_kernel(
    const float* __restrict__ in, bf16_t* __restrict__ out, int n) {
  int i = blockIdx.x * 256 + threadIdx.x;
  if (i < n) out[i] = (bf16_t)in[i];
}

// ---------------------------------------------------------------------------
// Embedding * sqrt(D) + sinusoidal positional encoding  -> h (f32 [M,D])
// ---------------------------------------------------------------------------
__global__ __launch_bounds__(256) void embed_kernel(
    const int* __restrict__ x, const float* __restrict__ emb,
    float* __restrict__ h) {
  int i = blockIdx.x * 256 + threadIdx.x;
  if (i >= MTOT * DD) return;
  int m = i / DD, d = i - m * DD;
  int spos = m & (SS - 1);
  int tok  = x[m];
  float val = emb[(size_t)tok * DD + d] * 27.712812921102035f; // sqrt(768)
  float div = __expf(-9.210340371976184f * (float)(d & ~1) * (1.0f / 768.0f));
  float ang = (float)spos * div;
  val += (d & 1) ? __cosf(ang) : __sinf(ang);
  h[i] = val;
}

// ---------------------------------------------------------------------------
// LayerNorm over D=768, one block (256 threads) per row, bf16 output
// ---------------------------------------------------------------------------
__global__ __launch_bounds__(256) void layernorm_kernel(
    const float* __restrict__ x, const float* __restrict__ g,
    const float* __restrict__ b, bf16_t* __restrict__ out) {
  __shared__ float red[256];
  const int row = blockIdx.x, tid = threadIdx.x;
  const float* xr = x + (size_t)row * DD;
  float v0 = xr[tid], v1 = xr[tid + 256], v2 = xr[tid + 512];
  red[tid] = v0 + v1 + v2;
  __syncthreads();
  for (int o = 128; o > 0; o >>= 1) {
    if (tid < o) red[tid] += red[tid + o];
    __syncthreads();
  }
  float mu = red[0] * (1.0f / DD);
  __syncthreads();
  float d0 = v0 - mu, d1 = v1 - mu, d2 = v2 - mu;
  red[tid] = d0 * d0 + d1 * d1 + d2 * d2;
  __syncthreads();
  for (int o = 128; o > 0; o >>= 1) {
    if (tid < o) red[tid] += red[tid + o];
    __syncthreads();
  }
  float rstd = rsqrtf(red[0] * (1.0f / DD) + 1e-5f);
  bf16_t* orow = out + (size_t)row * DD;
  orow[tid]       = (bf16_t)(d0 * rstd * g[tid]       + b[tid]);
  orow[tid + 256] = (bf16_t)(d1 * rstd * g[tid + 256] + b[tid + 256]);
  orow[tid + 512] = (bf16_t)(d2 * rstd * g[tid + 512] + b[tid + 512]);
}

// ---------------------------------------------------------------------------
// Tiled bf16 WMMA GEMM:  C[M,N] = A[M,K] @ W[K,N] + bias (+relu) (+residual)
// Block = 256 threads (8 waves), tile 128x128, K-slab 32.
// Wave grid 4x2 -> each wave computes 32x64 (2x4 WMMA subtiles).
// LDS: A K-major [128][32] (async DMA when available), W N-major [128n][32k].
// Next-slab lines prefetched (global_prefetch_b8) behind the 8 WMMAs.
// ---------------------------------------------------------------------------
template <int RELU, int OUT_BF16, int RESIDUAL>
__global__ __launch_bounds__(256) void gemm_bf16_kernel(
    const bf16_t* __restrict__ A, const bf16_t* __restrict__ W,
    const float* __restrict__ bias, const float* __restrict__ resid,
    float* __restrict__ outf, bf16_t* __restrict__ outb,
    int M, int N, int K) {
  __shared__ alignas(16) bf16_t ldsA[128 * 32];
  __shared__ alignas(16) bf16_t ldsB[128 * 32];

  const int tid  = threadIdx.x;
  const int lane = tid & 31, wave = tid >> 5;
  const int wm = wave >> 1, wn = wave & 1;
  const int l15 = lane & 15, lhi = lane >> 4;
  const int brow = blockIdx.y * 128, bcol = blockIdx.x * 128;

  v8f acc[2][4] = {};

  for (int kb = 0; kb < K; kb += 32) {
    __syncthreads();
    // Stage A tile: 128 rows x 32 K (512 x b128 / 256 threads)
#pragma unroll
    for (int t = 0; t < 2; ++t) {
      int idx = tid + t * 256;
      int row = idx >> 2, kq = (idx & 3) * 8;
      const bf16_t* src = A + (size_t)(brow + row) * K + kb + kq;
#if USE_ASYNC_LDS
      ASYNC_COPY_B128(src, ldsA + row * 32 + kq);
#else
      *(uint4*)(ldsA + row * 32 + kq) = *(const uint4*)src;
#endif
    }
    // Stage W tile transposed: global [k][n] -> LDS [n][k]
#pragma unroll
    for (int t = 0; t < 2; ++t) {
      int idx = tid + t * 256;
      int kr = idx >> 4, nq = (idx & 15) * 8;
      Pack8 p;
      p.u = *(const uint4*)(W + (size_t)(kb + kr) * N + bcol + nq);
      unsigned short* bs = (unsigned short*)ldsB;
#pragma unroll
      for (int e = 0; e < 8; ++e) bs[(nq + e) * 32 + kr] = p.us[e];
    }
#if USE_ASYNC_LDS
    ASYNC_WAIT();
#endif
    __syncthreads();

    // Prefetch next K-slab lines into cache behind this slab's WMMAs
    if (kb + 32 < K) {
#pragma unroll
      for (int t = 0; t < 2; ++t) {
        int idx = tid + t * 256;
        int row = idx >> 2, kq = (idx & 3) * 8;
        __builtin_prefetch(A + (size_t)(brow + row) * K + kb + 32 + kq, 0, 1);
        int kr = idx >> 4, nq = (idx & 15) * 8;
        __builtin_prefetch(W + (size_t)(kb + 32 + kr) * N + bcol + nq, 0, 1);
      }
    }

    v16bf af[2], bf[4];
#pragma unroll
    for (int mi = 0; mi < 2; ++mi)
      af[mi] = load_frag_a(ldsA + (wm * 32 + mi * 16 + l15) * 32, lane);
#pragma unroll
    for (int ni = 0; ni < 4; ++ni)
      bf[ni] = load_frag_b(ldsB + (wn * 64 + ni * 16 + l15) * 32, lane);
#pragma unroll
    for (int mi = 0; mi < 2; ++mi)
#pragma unroll
      for (int ni = 0; ni < 4; ++ni)
        acc[mi][ni] = wmma_bf16(af[mi], bf[ni], acc[mi][ni]);
  }

  // Epilogue. C layout: elem r -> row = r + 8*lhi, col = lane%16.
#pragma unroll
  for (int mi = 0; mi < 2; ++mi)
#pragma unroll
    for (int ni = 0; ni < 4; ++ni)
#pragma unroll
      for (int r = 0; r < 8; ++r) {
        int row = brow + wm * 32 + mi * 16 + r + lhi * 8;
        int col = bcol + wn * 64 + ni * 16 + l15;
        float v = acc[mi][ni][r] + bias[col];
        if (RELU) v = fmaxf(v, 0.0f);
        if (RESIDUAL) v += resid[(size_t)row * N + col];
        if (OUT_BF16) outb[(size_t)row * N + col] = (bf16_t)v;
        else          outf[(size_t)row * N + col] = v;
      }
  (void)outf; (void)outb; (void)resid;
}

// ---------------------------------------------------------------------------
// Flash attention, causal. qkv bf16 [M, 3D]; out bf16 [M, D].
// Block = 128 threads (4 waves) handles 64 query rows of one (b,h);
// each wave owns 16 query rows. Key blocks of 32; online softmax.
// K tile staged via async global->LDS DMA when available; V transposed
// through VGPRs. Row statistics reduced over 16-lane half-groups (wave32
// shfl_xor -> ds_bpermute).
// ---------------------------------------------------------------------------
__global__ __launch_bounds__(128) void attn_kernel(
    const bf16_t* __restrict__ qkv, bf16_t* __restrict__ outp) {
  __shared__ alignas(16) bf16_t ldsK[32 * 64];    // [key][dim] K-major
  __shared__ alignas(16) bf16_t ldsVT[64 * 32];   // [outdim][key]
  __shared__ alignas(16) bf16_t ldsP[4][16 * 32]; // per-wave P tile

  const int tid = threadIdx.x, lane = tid & 31, wave = tid >> 5;
  const int l15 = lane & 15, lhi = lane >> 4;
  const int qt = blockIdx.x, bh = blockIdx.y;
  const int bb = bh / HH, hh = bh - bb * HH;
  const int q0 = bb * SS + qt * 64;

  // Q fragments (row = q0 + wave*16 + lane%16), two 32-dim chunks
  v16bf qf[2];
  {
    const bf16_t* qb =
        qkv + (size_t)(q0 + wave * 16 + l15) * (3 * DD) + hh * HDIM;
    qf[0] = load_frag_a(qb, lane);
    qf[1] = load_frag_a(qb + 32, lane);
  }

  v8f O[4] = {};
  float mrow[8], lrow[8];
#pragma unroll
  for (int r = 0; r < 8; ++r) { mrow[r] = -1e30f; lrow[r] = 0.0f; }

  const int nblk = 2 * qt + 2; // keys up to (qt+1)*64 cover the causal span
  for (int jb = 0; jb < nblk; ++jb) {
    const int kbase = jb * 32;
    __syncthreads();
    // Stage K tile (async DMA, K-major) and V tile (transposed via VGPRs)
#pragma unroll
    for (int t = 0; t < 2; ++t) {
      int idx = tid + t * 128;
      int kr = idx >> 3, kq = (idx & 7) * 8;
      size_t rbase = (size_t)(bb * SS + kbase + kr) * (3 * DD) + hh * HDIM + kq;
#if USE_ASYNC_LDS
      ASYNC_COPY_B128(qkv + rbase + DD, ldsK + kr * 64 + kq);
#else
      *(uint4*)(ldsK + kr * 64 + kq) = *(const uint4*)(qkv + rbase + DD);
#endif
      Pack8 pv;
      pv.u = *(const uint4*)(qkv + rbase + 2 * DD);
      unsigned short* vs = (unsigned short*)ldsVT;
#pragma unroll
      for (int e = 0; e < 8; ++e) vs[(kq + e) * 32 + kr] = pv.us[e];
      // Prefetch next key-block lines
      if (jb + 1 < nblk) {
        size_t nbase =
            (size_t)(bb * SS + kbase + 32 + kr) * (3 * DD) + hh * HDIM + kq;
        __builtin_prefetch(qkv + nbase + DD, 0, 1);
        __builtin_prefetch(qkv + nbase + 2 * DD, 0, 1);
      }
    }
#if USE_ASYNC_LDS
    ASYNC_WAIT();
#endif
    __syncthreads();

    // Scores: S = Q @ K^T for 2 key subtiles of 16
    v8f s[2] = {};
#pragma unroll
    for (int sub = 0; sub < 2; ++sub)
#pragma unroll
      for (int ch = 0; ch < 2; ++ch) {
        v16bf kf = load_frag_b(ldsK + (sub * 16 + l15) * 64 + ch * 32, lane);
        s[sub] = wmma_bf16(qf[ch], kf, s[sub]);
      }

    // Scale, causal mask, online softmax update, stash P to LDS (bf16)
#pragma unroll
    for (int r = 0; r < 8; ++r) {
      int qrow = qt * 64 + wave * 16 + r + lhi * 8;
      float sv0 = s[0][r] * 0.125f;
      float sv1 = s[1][r] * 0.125f;
      if (kbase + 0  + l15 > qrow) sv0 = -1e30f;
      if (kbase + 16 + l15 > qrow) sv1 = -1e30f;
      float mb = fmaxf(sv0, sv1);
#pragma unroll
      for (int off = 8; off >= 1; off >>= 1)
        mb = fmaxf(mb, __shfl_xor(mb, off, 32));
      float mnew  = fmaxf(mrow[r], mb);
      float alpha = __expf(mrow[r] - mnew);
      float p0 = __expf(sv0 - mnew);
      float p1 = __expf(sv1 - mnew);
      float sb = p0 + p1;
#pragma unroll
      for (int off = 8; off >= 1; off >>= 1)
        sb += __shfl_xor(sb, off, 32);
      lrow[r] = lrow[r] * alpha + sb;
      mrow[r] = mnew;
#pragma unroll
      for (int c = 0; c < 4; ++c) O[c][r] *= alpha;
      int prow = r + lhi * 8;
      ldsP[wave][prow * 32 + l15]      = (bf16_t)p0;
      ldsP[wave][prow * 32 + 16 + l15] = (bf16_t)p1;
    }

    // O += P @ V (per-wave LDS, wave-order LDS keeps this consistent)
    v16bf pf = load_frag_a(ldsP[wave] + l15 * 32, lane);
#pragma unroll
    for (int c = 0; c < 4; ++c) {
      v16bf vf = load_frag_b(ldsVT + (c * 16 + l15) * 32, lane);
      O[c] = wmma_bf16(pf, vf, O[c]);
    }
  }

  // Normalize and write bf16 output
#pragma unroll
  for (int c = 0; c < 4; ++c)
#pragma unroll
    for (int r = 0; r < 8; ++r) {
      int row = q0 + wave * 16 + r + lhi * 8;
      int col = hh * HDIM + c * 16 + l15;
      outp[(size_t)row * DD + col] = (bf16_t)(O[c][r] / lrow[r]);
    }
}

// ---------------------------------------------------------------------------
// Host-side orchestration
// ---------------------------------------------------------------------------
extern "C" void kernel_launch(void* const* d_in, const int* in_sizes, int n_in,
                              void* d_out, int out_size, void* d_ws,
                              size_t ws_size, hipStream_t stream) {
  (void)in_sizes; (void)n_in; (void)out_size; (void)ws_size;

  const int*   x      = (const int*)d_in[0];
  const float* emb    = (const float*)d_in[1];
  const float* qkv_w  = (const float*)d_in[2];
  const float* qkv_b  = (const float*)d_in[3];
  const float* out_w  = (const float*)d_in[4];
  const float* out_b  = (const float*)d_in[5];
  const float* ln1_s  = (const float*)d_in[6];
  const float* ln1_b  = (const float*)d_in[7];
  const float* w1     = (const float*)d_in[8];
  const float* b1     = (const float*)d_in[9];
  const float* w2     = (const float*)d_in[10];
  const float* b2     = (const float*)d_in[11];
  const float* ln2_s  = (const float*)d_in[12];
  const float* ln2_b  = (const float*)d_in[13];
  const float* lnf_s  = (const float*)d_in[14];
  const float* lnf_b  = (const float*)d_in[15];
  const float* head_w = (const float*)d_in[16];
  const float* head_b = (const float*)d_in[17];

  // Workspace layout (bytes, 256-aligned)
  char* ws = (char*)d_ws;
  size_t off = 0;
  auto take = [&](size_t bytes) -> char* {
    char* p = ws + off;
    off = (off + bytes + 255) & ~(size_t)255;
    return p;
  };
  const size_t n_wq = (size_t)LLN * DD * 3 * DD;
  const size_t n_wo = (size_t)LLN * DD * DD;
  const size_t n_w1 = (size_t)LLN * DD * FFD;
  const size_t n_w2 = (size_t)LLN * FFD * DD;
  const size_t n_wh = (size_t)DD * VV;

  bf16_t* wq_bf  = (bf16_t*)take(n_wq * 2);
  bf16_t* wo_bf  = (bf16_t*)take(n_wo * 2);
  bf16_t* w1_bf  = (bf16_t*)take(n_w1 * 2);
  bf16_t* w2_bf  = (bf16_t*)take(n_w2 * 2);
  bf16_t* wh_bf  = (bf16_t*)take(n_wh * 2);
  float*  h      = (float*)take((size_t)MTOT * DD * 4);
  bf16_t* act16  = (bf16_t*)take((size_t)MTOT * DD * 2);
  bf16_t* big    = (bf16_t*)take((size_t)MTOT * FFD * 2); // qkv & ffn1 share

  auto cvt = [&](const float* src, bf16_t* dst, size_t n) {
    cvt_bf16_kernel<<<(unsigned)((n + 255) / 256), 256, 0, stream>>>(
        src, dst, (int)n);
  };
  cvt(qkv_w, wq_bf, n_wq);
  cvt(out_w, wo_bf, n_wo);
  cvt(w1, w1_bf, n_w1);
  cvt(w2, w2_bf, n_w2);
  cvt(head_w, wh_bf, n_wh);

  // h = emb[x]*sqrt(D) + PE
  embed_kernel<<<(MTOT * DD + 255) / 256, 256, 0, stream>>>(x, emb, h);

  const int MB = MTOT / 128; // 64 row-tiles
  for (int l = 0; l < LLN; ++l) {
    // act16 = LN1(h)
    layernorm_kernel<<<MTOT, 256, 0, stream>>>(h, ln1_s + l * DD,
                                               ln1_b + l * DD, act16);
    // big = act16 @ qkv_w + qkv_b   (bf16 out, [M, 3D])
    gemm_bf16_kernel<0, 1, 0><<<dim3(3 * DD / 128, MB), 256, 0, stream>>>(
        act16, wq_bf + (size_t)l * DD * 3 * DD, qkv_b + (size_t)l * 3 * DD,
        nullptr, nullptr, big, MTOT, 3 * DD, DD);
    // act16 = attention(big)
    attn_kernel<<<dim3(SS / 64, BB * HH), 128, 0, stream>>>(big, act16);
    // h += act16 @ out_w + out_b
    gemm_bf16_kernel<0, 0, 1><<<dim3(DD / 128, MB), 256, 0, stream>>>(
        act16, wo_bf + (size_t)l * DD * DD, out_b + (size_t)l * DD, h, h,
        nullptr, MTOT, DD, DD);
    // act16 = LN2(h)
    layernorm_kernel<<<MTOT, 256, 0, stream>>>(h, ln2_s + l * DD,
                                               ln2_b + l * DD, act16);
    // big = relu(act16 @ w1 + b1)   (bf16 out, [M, F])
    gemm_bf16_kernel<1, 1, 0><<<dim3(FFD / 128, MB), 256, 0, stream>>>(
        act16, w1_bf + (size_t)l * DD * FFD, b1 + (size_t)l * FFD, nullptr,
        nullptr, big, MTOT, FFD, DD);
    // h += big @ w2 + b2
    gemm_bf16_kernel<0, 0, 1><<<dim3(DD / 128, MB), 256, 0, stream>>>(
        big, w2_bf + (size_t)l * FFD * DD, b2 + (size_t)l * DD, h, h, nullptr,
        MTOT, DD, FFD);
  }

  // Final LN + head projection -> d_out f32 [M, V]
  layernorm_kernel<<<MTOT, 256, 0, stream>>>(h, lnf_s, lnf_b, act16);
  gemm_bf16_kernel<0, 0, 0><<<dim3(VV / 128, MB), 256, 0, stream>>>(
      act16, wh_bf, head_b, nullptr, (float*)d_out, nullptr, MTOT, VV, DD);
}
